// DropScaledDotProductAttention_23854248362055
// MI455X (gfx1250) — matrix operand order repaired and checked by reading, not compiled, not run on hardware
//
#include <hip/hip_runtime.h>

// DropScaledDotProductAttention for MI455X (gfx1250), wave32 + WMMA.
// B=2, N=16, LQ=512, DK=DV=64, DD=128, H2=256. TEMPERATURE=8.

#define BB   2
#define NH   16
#define LQ   512
#define DK   64
#define DV   64
#define H2   256
#define NEG_INF (-1e9f)

#define H2P  (H2 + 4)    // padded LDS row (bank spread)
#define LQP  (LQ + 4)
#define DVP  (DV + 4)

typedef _Float16 v16h __attribute__((ext_vector_type(16)));
typedef float    v8f  __attribute__((ext_vector_type(8)));

// ---------------------------------------------------------------------------
// Kernel 1: A[b,i,g] = d0[b,i,:]·W1[0:128,g] + b1[g]
//           Bm[b,j,g] = d1[b,j,:]·W1[128:256,g]
// One wave per 16x16 output tile; WMMA f32_16x16x32_f16, K-loop over 128.
// grid = (1024/16, 256/16, 2), block = 32.
// ---------------------------------------------------------------------------
__global__ __launch_bounds__(32)
void mlp_in_gemm(const float* __restrict__ d0, const float* __restrict__ d1,
                 const float* __restrict__ W1, const float* __restrict__ b1,
                 float* __restrict__ Aout, float* __restrict__ Bout)
{
    const int lane = threadIdx.x;        // 0..31
    const int half = lane >> 4;          // 0 | 1
    const int r    = lane & 15;
    const int m0   = blockIdx.x * 16;    // flat row (b*LQ+i) tile
    const int n0   = blockIdx.y * 16;    // g tile
    const int z    = blockIdx.z;         // 0: A-path, 1: B-path

    const float* src = z ? d1 : d0;                  // [1024,128]
    const float* w   = W1 + (z ? 128 * H2 : 0);      // [128,256] half
    float*       out = z ? Bout : Aout;              // [1024,256]

    v8f acc = {};
    for (int kb = 0; kb < 128; kb += 32) {
        v16h af, bf;
        const int arow = m0 + r;
        #pragma unroll
        for (int e = 0; e < 16; ++e) {
            // A frag: lane holds row M=r, K = kb + half*8 + (e<8?e:e+8)
            int ka = kb + half * 8 + (e < 8 ? e : e + 8);
            af[e] = (_Float16)src[arow * 128 + ka];
            // B frag: lane holds col N=r, K = kb + half*8 + (e<8?e:e+8)
            int kk = kb + half * 8 + (e < 8 ? e : e + 8);
            bf[e] = (_Float16)w[kk * H2 + n0 + r];
        }
        acc = __builtin_amdgcn_wmma_f32_16x16x32_f16(
            false, af, false, bf, (short)0, acc, false, false);
    }

    const float bias = z ? 0.0f : b1[n0 + r];
    #pragma unroll
    for (int rr = 0; rr < 8; ++rr) {
        int row = m0 + half * 8 + rr;     // C/D: VGPR rr -> M = rr + half*8
        out[row * H2 + n0 + r] = acc[rr] + bias;
    }
}

// ---------------------------------------------------------------------------
// Kernel 2: decisions[b,i,j] = (sum_g relu(A[b,i,g]+Bm[b,j,g]) * w[g] + c0) > 0
// where w[g] = W2[g,1]-W2[g,0], c0 = b2[1]-b2[0]  (argmax==1 <=> strict >)
// Writes reference-layout dec[b][i][j] (d_out) and transposed decT[b][j][i]
// (workspace, consumed by attention with vector loads).
// grid = (32, 32, 2), block = 256; padded LDS rows kill 16-way bank conflicts.
// ---------------------------------------------------------------------------
__global__ __launch_bounds__(256)
void decide(const float* __restrict__ Aact, const float* __restrict__ Bact,
            const float* __restrict__ W2, const float* __restrict__ b2,
            float* __restrict__ dec, float* __restrict__ decT)
{
    __shared__ float sA[16][H2P];
    __shared__ float sB[16][H2P];
    __shared__ float sw[H2];

    const int t  = threadIdx.x;
    const int bI = blockIdx.x, bJ = blockIdx.y, b = blockIdx.z;

    #pragma unroll 4
    for (int qr = 0; qr < 16; ++qr) {
        sA[qr][t] = Aact[(b * LQ + bI * 16 + qr) * H2 + t];
        sB[qr][t] = Bact[(b * LQ + bJ * 16 + qr) * H2 + t];
    }
    sw[t] = W2[2 * t + 1] - W2[2 * t];
    __syncthreads();

    const float c0 = b2[1] - b2[0];
    const int ti = t & 15;        // i within tile
    const int tj = t >> 4;        // j within tile

    // float4 LDS reads: rows 16B-aligned (H2P*4 = 1040 bytes), banks spread
    const float4* rA = (const float4*)&sA[ti][0];
    const float4* rB = (const float4*)&sB[tj][0];
    const float4* rw = (const float4*)&sw[0];
    float acc = c0;
    #pragma unroll 4
    for (int g4 = 0; g4 < H2 / 4; ++g4) {
        float4 a = rA[g4], bb = rB[g4], ww = rw[g4];
        acc = fmaf(fmaxf(a.x + bb.x, 0.0f), ww.x, acc);
        acc = fmaf(fmaxf(a.y + bb.y, 0.0f), ww.y, acc);
        acc = fmaf(fmaxf(a.z + bb.z, 0.0f), ww.z, acc);
        acc = fmaf(fmaxf(a.w + bb.w, 0.0f), ww.w, acc);
    }
    const float d = (acc > 0.0f) ? 1.0f : 0.0f;
    const int ig = bI * 16 + ti, jg = bJ * 16 + tj;
    dec [(size_t)(b * LQ + ig) * LQ + jg] = d;   // reference layout [b,1,i,j]
    decT[(size_t)(b * LQ + jg) * LQ + ig] = d;   // transposed for attention
}

// ---------------------------------------------------------------------------
// Kernel 3: per (head, 16-row i-tile): S = mask(q/8 @ k^T), softmax rows,
// attn out, O = P @ v. grid = (LQ/16, B*NH), block = 512 (16 waves).
// ---------------------------------------------------------------------------
__global__ __launch_bounds__(512)
void attention(const float* __restrict__ q, const float* __restrict__ k,
               const float* __restrict__ v, const float* __restrict__ decT,
               float* __restrict__ outO, float* __restrict__ outP)
{
    __shared__ float S[16][LQP];    // scores, then probabilities (padded rows)
    __shared__ float Ot[16][DVP];   // output accumulator (padded rows)

    const int t    = threadIdx.x;
    const int lane = t & 31;
    const int wv   = t >> 5;          // wave 0..15
    const int half = lane >> 4;
    const int r    = lane & 15;
    const int it   = blockIdx.x;      // i-tile 0..31
    const int bh   = blockIdx.y;      // b*NH+n
    const int b    = bh >> 4;

    // zero output accumulator (consumed after the softmax barrier)
    for (int idx = t; idx < 16 * DVP; idx += 512) ((float*)Ot)[idx] = 0.0f;

    // q A-fragments for K=0..31 and K=32..63, pre-scaled by 1/TEMPERATURE
    v16h aq0, aq1;
    {
        const int row = it * 16 + r;
        const float* qp = q + (size_t)(bh * LQ + row) * DK;
        #pragma unroll
        for (int e = 0; e < 16; ++e) {
            int kl = half * 8 + (e < 8 ? e : e + 8);
            aq0[e] = (_Float16)(qp[kl] * 0.125f);
            aq1[e] = (_Float16)(qp[32 + kl] * 0.125f);
        }
    }

    // ---- masked scores: each wave does 2 of the 32 j-tiles ----
    for (int jt = wv; jt < 32; jt += 16) {
        v16h bk0, bk1;
        const int col = jt * 16 + r;
        const float* kp = k + (size_t)(bh * LQ + col) * DK;
        #pragma unroll
        for (int e = 0; e < 16; ++e) {
            int kd = half * 8 + (e < 8 ? e : e + 8);
            bk0[e] = (_Float16)kp[kd];
            bk1[e] = (_Float16)kp[32 + kd];
        }
        v8f c = {};
        c = __builtin_amdgcn_wmma_f32_16x16x32_f16(false, aq0, false, bk0,
                                                   (short)0, c, false, false);
        c = __builtin_amdgcn_wmma_f32_16x16x32_f16(false, aq1, false, bk1,
                                                   (short)0, c, false, false);

        // mask: lane has fixed j=col, 8 consecutive i -> two float4 loads
        const int jg = jt * 16 + r;
        const float4* dp =
            (const float4*)&decT[(size_t)(b * LQ + jg) * LQ + it * 16 + half * 8];
        const float4 dA = dp[0], dB = dp[1];
        const float dm[8] = { dA.x, dA.y, dA.z, dA.w, dB.x, dB.y, dB.z, dB.w };
        #pragma unroll
        for (int rr = 0; rr < 8; ++rr) {
            int iloc = half * 8 + rr;
            S[iloc][jg] = (dm[rr] > 0.5f) ? c[rr] : NEG_INF;
        }
    }
    __syncthreads();

    // ---- softmax: wave wv owns row wv; wave32 shuffle reduction ----
    {
        const int row = wv;
        float mx = -3.4e38f;
        for (int cc = lane; cc < LQ; cc += 32) mx = fmaxf(mx, S[row][cc]);
        #pragma unroll
        for (int off = 16; off; off >>= 1) mx = fmaxf(mx, __shfl_xor(mx, off, 32));
        float sum = 0.0f;
        for (int cc = lane; cc < LQ; cc += 32) {
            float p = __expf(S[row][cc] - mx);
            S[row][cc] = p;
            sum += p;
        }
        #pragma unroll
        for (int off = 16; off; off >>= 1) sum += __shfl_xor(sum, off, 32);
        const float inv = 1.0f / sum;
        const int ig = it * 16 + row;
        float* pp = outP + (size_t)(bh * LQ + ig) * LQ;
        for (int cc = lane; cc < LQ; cc += 32) {
            float p = S[row][cc] * inv;
            S[row][cc] = p;
            pp[cc] = p;
        }
    }
    __syncthreads();

    // ---- O = P @ v : wave wv owns K chunk [wv*32, wv*32+32) ----
    {
        v16h ap;
        #pragma unroll
        for (int e = 0; e < 16; ++e) {
            int kl = half * 8 + (e < 8 ? e : e + 8);
            ap[e] = (_Float16)S[r][wv * 32 + kl];   // padded rows: banks spread
        }
        #pragma unroll
        for (int nt = 0; nt < 4; ++nt) {
            v16h bv;
            #pragma unroll
            for (int e = 0; e < 16; ++e) {
                int jg = wv * 32 + half * 8 + (e < 8 ? e : e + 8);
                bv[e] = (_Float16)v[(size_t)(bh * LQ + jg) * DV + nt * 16 + r];
            }
            v8f c = {};
            c = __builtin_amdgcn_wmma_f32_16x16x32_f16(false, ap, false, bv,
                                                       (short)0, c, false, false);
            #pragma unroll
            for (int rr = 0; rr < 8; ++rr)
                atomicAdd(&Ot[half * 8 + rr][nt * 16 + r], c[rr]);  // ds_add_f32
        }
    }
    __syncthreads();

    // write 16x64 output tile
    for (int idx = t; idx < 16 * DV; idx += 512) {
        int row = idx >> 6, col = idx & 63;
        outO[(size_t)(bh * LQ + it * 16 + row) * DV + col] = Ot[row][col];
    }
}

// ---------------------------------------------------------------------------
extern "C" void kernel_launch(void* const* d_in, const int* in_sizes, int n_in,
                              void* d_out, int out_size, void* d_ws, size_t ws_size,
                              hipStream_t stream)
{
    (void)in_sizes; (void)n_in; (void)out_size; (void)ws_size;

    const float* q  = (const float*)d_in[0];
    const float* k  = (const float*)d_in[1];
    const float* v  = (const float*)d_in[2];
    const float* d0 = (const float*)d_in[3];
    const float* d1 = (const float*)d_in[4];
    const float* W1 = (const float*)d_in[5];
    const float* b1 = (const float*)d_in[6];
    const float* W2 = (const float*)d_in[7];
    const float* b2 = (const float*)d_in[8];

    float* outO = (float*)d_out;                         // [2,16,512,64]
    float* outP = outO + (size_t)BB * NH * LQ * DV;      // [2,16,512,512]
    float* outD = outP + (size_t)BB * NH * LQ * LQ;      // [2,1,512,512] as 0/1 f32

    float* Aact = (float*)d_ws;                          // [2*512, 256]  1 MB
    float* Bact = Aact + (size_t)BB * LQ * H2;           // [2*512, 256]  1 MB
    float* decT = Bact + (size_t)BB * LQ * H2;           // [2,512(j),512(i)] 2 MB

    // 1) factored MLP input GEMMs (WMMA)
    mlp_in_gemm<<<dim3((BB * LQ) / 16, H2 / 16, 2), 32, 0, stream>>>(
        d0, d1, W1, b1, Aact, Bact);

    // 2) pairwise decisions (LDS-tiled VALU; relu breaks further factoring)
    decide<<<dim3(LQ / 16, LQ / 16, BB), 256, 0, stream>>>(
        Aact, Bact, W2, b2, outD, decT);

    // 3) masked attention: QK^T (WMMA) -> softmax -> PV (WMMA)
    attention<<<dim3(LQ / 16, BB * NH), 512, 0, stream>>>(
        q, k, v, decT, outO, outP);
}
